// TemporalSequenceLearner_59828894433747
// MI455X (gfx1250) — compile-verified, matrix operand order checked by reading, not run on hardware
//
#include <hip/hip_runtime.h>
#include <math.h>

// ---------------------------------------------------------------------------
// Types for WMMA bf16 path (CDNA5 / gfx1250, wave32)
// ---------------------------------------------------------------------------
typedef __attribute__((ext_vector_type(16))) __bf16 bf16x16;
typedef __attribute__((ext_vector_type(8)))  __bf16 bf16x8;
typedef __attribute__((ext_vector_type(8)))  float  f32x8;

__device__ __forceinline__ bf16x16 cat8(bf16x8 lo, bf16x8 hi) {
  return __builtin_shufflevector(lo, hi, 0,1,2,3,4,5,6,7,8,9,10,11,12,13,14,15);
}

// ---------------------------------------------------------------------------
// Pack conv weights: fp32 OIHW [COUT][KEFF] -> bf16 [COUTP][KPAD] zero-padded
// (row-major, K contiguous: each lane loads 8-element runs at k0+8h and k0+8h+16)
// ---------------------------------------------------------------------------
__global__ void pack_conv_w(const float* __restrict__ w, __bf16* __restrict__ out,
                            int COUT, int KEFF, int KPAD, int COUTP) {
  int idx = blockIdx.x * blockDim.x + threadIdx.x;
  int total = COUTP * KPAD;
  if (idx >= total) return;
  int row = idx / KPAD;
  int k   = idx - row * KPAD;
  float v = (row < COUT && k < KEFF) ? w[(size_t)row * KEFF + k] : 0.0f;
  out[idx] = (__bf16)v;
}

// ---------------------------------------------------------------------------
// Pack RNN layer-1 weights (x-part, k<1152) into bf16 [192][1152]:
// rows = mat*48 + j  (mat: 0=ff1(masked),1=ff2(masked),2=time_a,3=time_b)
// ---------------------------------------------------------------------------
__global__ void pack_rnn_w(const float* __restrict__ ff1w, const float* __restrict__ ff2w,
                           const float* __restrict__ taw,  const float* __restrict__ tbw,
                           const int* __restrict__ mask, __bf16* __restrict__ out) {
  int idx = blockIdx.x * blockDim.x + threadIdx.x;
  if (idx >= 192 * 1152) return;
  int row = idx / 1152;
  int k   = idx - row * 1152;
  int mat = row / 48;
  int j   = row - mat * 48;
  float v = 0.0f;
  if (j < 37) {
    const float* w = (mat == 0) ? ff1w : (mat == 1) ? ff2w : (mat == 2) ? taw : tbw;
    v = w[j * 1189 + k];
    if (mat < 2) v *= (float)mask[j * 1189 + k];
  }
  out[idx] = (__bf16)v;
}

// ---------------------------------------------------------------------------
// Implicit-GEMM conv + bias + ReLU, bf16 WMMA, f32 accumulate.
// One wave computes a 16-position tile for ALL output-channel tiles (MT),
// reusing the gathered im2col B-fragment across M-tiles.
// grid = (ceil(OH*OW/16), 1, NIMG), block = 32.
// ---------------------------------------------------------------------------
template<int CIN,int H,int W,int KH,int KW,int STR,int COUT,int OH,int OW,
         int KPAD,int MT,bool INF32>
__global__ __launch_bounds__(32) void conv_wmma(
    const void* __restrict__ inv,
    const __bf16* __restrict__ wpack,   // [MT*16][KPAD] bf16
    const float*  __restrict__ bias,    // [COUT]
    __bf16* __restrict__ out)           // [NIMG][COUT][OH*OW]
{
  constexpr int KEFF = CIN * KH * KW;
  constexpr int P    = OH * OW;
  const int lane = threadIdx.x;
  const int loc  = lane & 15;
  const int hf   = lane >> 4;
  const int kb   = hf * 8;
  const int img  = blockIdx.z;

  int p = blockIdx.x * 16 + loc;
  const bool pval = (p < P);
  if (!pval) p = P - 1;
  const int oh = p / OW, ow = p - oh * OW;
  const int ih0 = oh * STR, iw0 = ow * STR;

  const float*  inF = (const float*)inv  + (size_t)img * CIN * H * W;
  const __bf16* inB = (const __bf16*)inv + (size_t)img * CIN * H * W;

  f32x8 acc[MT];
  #pragma unroll
  for (int m = 0; m < MT; ++m) acc[m] = (f32x8)0.0f;

  for (int k0 = 0; k0 < KPAD; k0 += 32) {
    // ---- gather B fragment (im2col column for output position p) ----
    bf16x16 bfrag;
    #pragma unroll
    for (int e = 0; e < 16; ++e) {
      int k = k0 + kb + (e < 8 ? e : e + 8);   // per-lane K mapping (ISA A/B layout)
      float v = 0.0f;
      if (k < KEFF) {
        int ic = k / (KH * KW);
        int r  = k - ic * (KH * KW);
        int kh = r / KW;
        int kw = r - kh * KW;
        size_t idx = (size_t)ic * H * W + (size_t)(ih0 + kh) * W + (iw0 + kw);
        v = INF32 ? inF[idx] : (float)inB[idx];
      }
      bfrag[e] = (__bf16)v;
    }
    // ---- A fragments (weights) + WMMA, one per M-tile ----
    #pragma unroll
    for (int m = 0; m < MT; ++m) {
      const __bf16* wrow = wpack + (size_t)(m * 16 + loc) * KPAD + k0 + kb;
      bf16x16 afrag = cat8(*(const bf16x8*)(wrow), *(const bf16x8*)(wrow + 16));
      acc[m] = __builtin_amdgcn_wmma_f32_16x16x32_bf16(
          false, afrag, false, bfrag, (short)0, acc[m], false, false);
    }
  }

  if (pval) {
    #pragma unroll
    for (int m = 0; m < MT; ++m) {
      #pragma unroll
      for (int v = 0; v < 8; ++v) {
        int oc = m * 16 + v + 8 * hf;    // C layout: lane holds col N, VGPR v holds row
        if (oc < COUT) {
          float r = acc[m][v] + bias[oc];
          r = r > 0.0f ? r : 0.0f;
          out[((size_t)img * COUT + oc) * P + p] = (__bf16)r;
        }
      }
    }
  }
}

// ---------------------------------------------------------------------------
// feats[b*16+t][f] = mean over 3 views of conv5 output (bf16), f = c*18+w
// ---------------------------------------------------------------------------
__global__ void feats_mean(const __bf16* __restrict__ a5, __bf16* __restrict__ feats) {
  int idx = blockIdx.x * blockDim.x + threadIdx.x;
  if (idx >= 256 * 1152) return;
  int row = idx / 1152;           // b*16 + t
  int f   = idx - row * 1152;
  float s = 0.0f;
  #pragma unroll
  for (int v = 0; v < 3; ++v)
    s += (float)a5[(size_t)(row * 3 + v) * 1152 + f];
  feats[idx] = (__bf16)(s * (1.0f / 3.0f));
}

// ---------------------------------------------------------------------------
// Dense bf16 GEMM: C[M][ldc] = A[M][K] * B^T (B stored [N][K] row-major).
// Used for the parallel x-part precompute of the CfC layer-1 cell.
// grid = (N/16, M/16), block = 32.
// ---------------------------------------------------------------------------
__global__ __launch_bounds__(32) void gemm_bf16_tn(
    const __bf16* __restrict__ A, const __bf16* __restrict__ B,
    float* __restrict__ C, int K, int ldc)
{
  const int lane = threadIdx.x;
  const int loc  = lane & 15;
  const int hf   = lane >> 4;
  const int kb   = hf * 8;
  const __bf16* arow = A + (size_t)(blockIdx.y * 16 + loc) * K;
  const __bf16* brow = B + (size_t)(blockIdx.x * 16 + loc) * K;
  f32x8 acc = (f32x8)0.0f;
  for (int k0 = 0; k0 < K; k0 += 32) {
    bf16x16 af = cat8(*(const bf16x8*)(arow + k0 + kb), *(const bf16x8*)(arow + k0 + kb + 16));
    bf16x16 bf = cat8(*(const bf16x8*)(brow + k0 + kb), *(const bf16x8*)(brow + k0 + kb + 16));
    acc = __builtin_amdgcn_wmma_f32_16x16x32_bf16(false, af, false, bf, (short)0, acc, false, false);
  }
  int n = blockIdx.x * 16 + loc;
  #pragma unroll
  for (int v = 0; v < 8; ++v) {
    int m = blockIdx.y * 16 + v + 8 * hf;
    C[(size_t)m * ldc + n] = acc[v];
  }
}

// ---------------------------------------------------------------------------
// Serial CfC scan: one persistent 640-thread block, all recurrent weights in
// LDS, x-part contributions read from precomputed P. Fused output head.
// ---------------------------------------------------------------------------
struct RnnArgs {
  const float* w1[4]; const float* b1[4];
  const float* w2[4]; const float* b2[4];
  const float* w3[4]; const float* b3[4];
  const int *m1, *m2, *m3;
  const float *ow, *ob;
  const float *P;            // [256][192]
  float *out;                // [16][3]
};

__device__ __forceinline__ float sigmoidf_(float x) { return 1.0f / (1.0f + expf(-x)); }

__global__ __launch_bounds__(640) void rnn_serial(RnnArgs a) {
  __shared__ float sW1[4][37][37];
  __shared__ float sW2[4][24][61];
  __shared__ float sW3[4][3][27];
  __shared__ float sB1[4][37], sB2[4][24], sB3[4][3];
  __shared__ float h1[16][37], h2[16][24], h3[16][3];
  __shared__ float n1[16][37], n2[16][24], n3[16][3];
  __shared__ float sOw[9], sOb[3];
  const int tid = threadIdx.x;
  const int NT = 640;

  for (int i = tid; i < 4*37*37; i += NT) {
    int mat = i / (37*37), r = i % (37*37), j = r / 37, kk = r % 37;
    float w = a.w1[mat][j*1189 + 1152 + kk];
    if (mat < 2) w *= (float)a.m1[j*1189 + 1152 + kk];
    sW1[mat][j][kk] = w;
  }
  for (int i = tid; i < 4*24*61; i += NT) {
    int mat = i / (24*61), r = i % (24*61), j = r / 61, kk = r % 61;
    float w = a.w2[mat][j*61 + kk];
    if (mat < 2) w *= (float)a.m2[j*61 + kk];
    sW2[mat][j][kk] = w;
  }
  for (int i = tid; i < 4*3*27; i += NT) {
    int mat = i / (3*27), r = i % (3*27), j = r / 27, kk = r % 27;
    float w = a.w3[mat][j*27 + kk];
    if (mat < 2) w *= (float)a.m3[j*27 + kk];
    sW3[mat][j][kk] = w;
  }
  for (int i = tid; i < 4*37; i += NT) sB1[i/37][i%37] = a.b1[i/37][i%37];
  for (int i = tid; i < 4*24; i += NT) sB2[i/24][i%24] = a.b2[i/24][i%24];
  for (int i = tid; i < 4*3;  i += NT) sB3[i/3][i%3]   = a.b3[i/3][i%3];
  if (tid < 9) sOw[tid] = a.ow[tid];
  if (tid < 3) sOb[tid] = a.ob[tid];
  for (int i = tid; i < 16*37; i += NT) h1[i/37][i%37] = 0.0f;
  for (int i = tid; i < 16*24; i += NT) h2[i/24][i%24] = 0.0f;
  for (int i = tid; i < 16*3;  i += NT) h3[i/3][i%3]   = 0.0f;
  __syncthreads();

  for (int t = 0; t < 16; ++t) {
    // ---- layer 1 (inter, h=37): x-part from P, recurrent K=37 from LDS ----
    if (tid < 16*37) {
      int b = tid / 37, j = tid - b * 37;
      const float* prow = a.P + (size_t)(b * 16 + t) * 192;
      float a0 = prow[0*48 + j], a1 = prow[1*48 + j];
      float a2 = prow[2*48 + j], a3 = prow[3*48 + j];
      for (int kk = 0; kk < 37; ++kk) {
        float z = h1[b][kk];
        a0 += z * sW1[0][j][kk]; a1 += z * sW1[1][j][kk];
        a2 += z * sW1[2][j][kk]; a3 += z * sW1[3][j][kk];
      }
      float ff1 = tanhf(a0 + sB1[0][j]);
      float ff2 = tanhf(a1 + sB1[1][j]);
      float tg  = sigmoidf_((a2 + sB1[2][j]) + (a3 + sB1[3][j]));
      n1[b][j] = ff1 * (1.0f - tg) + tg * ff2;
    }
    __syncthreads();
    // ---- layer 2 (cmd, h=24): z = [h1_new(37), h2_old(24)] ----
    if (tid < 16*24) {
      int b = tid / 24, j = tid - b * 24;
      float a0 = 0, a1 = 0, a2 = 0, a3 = 0;
      for (int kk = 0; kk < 37; ++kk) {
        float z = n1[b][kk];
        a0 += z * sW2[0][j][kk]; a1 += z * sW2[1][j][kk];
        a2 += z * sW2[2][j][kk]; a3 += z * sW2[3][j][kk];
      }
      for (int kk = 0; kk < 24; ++kk) {
        float z = h2[b][kk];
        a0 += z * sW2[0][j][37+kk]; a1 += z * sW2[1][j][37+kk];
        a2 += z * sW2[2][j][37+kk]; a3 += z * sW2[3][j][37+kk];
      }
      float ff1 = tanhf(a0 + sB2[0][j]);
      float ff2 = tanhf(a1 + sB2[1][j]);
      float tg  = sigmoidf_((a2 + sB2[2][j]) + (a3 + sB2[3][j]));
      n2[b][j] = ff1 * (1.0f - tg) + tg * ff2;
    }
    __syncthreads();
    // ---- layer 3 (motor, h=3): z = [h2_new(24), h3_old(3)] ----
    if (tid < 16*3) {
      int b = tid / 3, j = tid - b * 3;
      float a0 = 0, a1 = 0, a2 = 0, a3 = 0;
      for (int kk = 0; kk < 24; ++kk) {
        float z = n2[b][kk];
        a0 += z * sW3[0][j][kk]; a1 += z * sW3[1][j][kk];
        a2 += z * sW3[2][j][kk]; a3 += z * sW3[3][j][kk];
      }
      for (int kk = 0; kk < 3; ++kk) {
        float z = h3[b][kk];
        a0 += z * sW3[0][j][24+kk]; a1 += z * sW3[1][j][24+kk];
        a2 += z * sW3[2][j][24+kk]; a3 += z * sW3[3][j][24+kk];
      }
      float ff1 = tanhf(a0 + sB3[0][j]);
      float ff2 = tanhf(a1 + sB3[1][j]);
      float tg  = sigmoidf_((a2 + sB3[2][j]) + (a3 + sB3[3][j]));
      n3[b][j] = ff1 * (1.0f - tg) + tg * ff2;
    }
    __syncthreads();
    for (int i = tid; i < 16*37; i += NT) h1[i/37][i%37] = n1[i/37][i%37];
    for (int i = tid; i < 16*24; i += NT) h2[i/24][i%24] = n2[i/24][i%24];
    for (int i = tid; i < 16*3;  i += NT) h3[i/3][i%3]   = n3[i/3][i%3];
    __syncthreads();
  }

  // ---- output head on final motor state ----
  if (tid < 16) {
    int b = tid;
    float y0 = sOb[0], y1 = sOb[1], y2 = sOb[2];
    for (int j = 0; j < 3; ++j) {
      float h = h3[b][j];
      y0 += h * sOw[0*3 + j];
      y1 += h * sOw[1*3 + j];
      y2 += h * sOw[2*3 + j];
    }
    a.out[b*3 + 0] = tanhf(y0);
    a.out[b*3 + 1] = sigmoidf_(y1);
    a.out[b*3 + 2] = sigmoidf_(y2);
  }
}

// ---------------------------------------------------------------------------
// Host launch
// ---------------------------------------------------------------------------
extern "C" void kernel_launch(void* const* d_in, const int* in_sizes, int n_in,
                              void* d_out, int out_size, void* d_ws, size_t ws_size,
                              hipStream_t stream) {
  (void)in_sizes; (void)n_in; (void)out_size; (void)ws_size;
  const float* images = (const float*)d_in[0];
  const float* cw[5]; const float* cb[5];
  for (int i = 0; i < 5; ++i) { cw[i] = (const float*)d_in[1 + i]; cb[i] = (const float*)d_in[6 + i]; }

  // Workspace carve-out (256B aligned blocks)
  size_t off = 0;
  auto take = [&](size_t bytes) { size_t o = off; off += (bytes + 255) & ~(size_t)255; return o; };
  char* ws = (char*)d_ws;

  static const int COUTv[5]  = {24, 36, 48, 64, 64};
  static const int COUTPv[5] = {32, 48, 48, 64, 64};
  static const int KEFFv[5]  = {75, 600, 900, 432, 576};
  static const int KPADv[5]  = {96, 608, 928, 448, 576};
  __bf16* wp[5];
  for (int i = 0; i < 5; ++i)
    wp[i] = (__bf16*)(ws + take((size_t)COUTPv[i] * KPADv[i] * 2));

  const size_t NIMG = 768;
  __bf16* a1 = (__bf16*)(ws + take(NIMG * 24 * 3038 * 2));
  __bf16* a2 = (__bf16*)(ws + take(NIMG * 36 * 658  * 2));
  __bf16* a3 = (__bf16*)(ws + take(NIMG * 48 * 110  * 2));
  __bf16* a4 = (__bf16*)(ws + take(NIMG * 64 * 60   * 2));
  __bf16* a5 = (__bf16*)(ws + take(NIMG * 64 * 18   * 2));
  __bf16* feats  = (__bf16*)(ws + take((size_t)256 * 1152 * 2));
  __bf16* wprnn  = (__bf16*)(ws + take((size_t)192 * 1152 * 2));
  float*  P      = (float*) (ws + take((size_t)256 * 192  * 4));

  // 1) Weight packing
  for (int i = 0; i < 5; ++i) {
    int total = COUTPv[i] * KPADv[i];
    pack_conv_w<<<(total + 255) / 256, 256, 0, stream>>>(cw[i], wp[i],
        COUTv[i], KEFFv[i], KPADv[i], COUTPv[i]);
  }
  pack_rnn_w<<<(192 * 1152 + 255) / 256, 256, 0, stream>>>(
      (const float*)d_in[11], (const float*)d_in[13],
      (const float*)d_in[15], (const float*)d_in[17],
      (const int*)d_in[37], wprnn);

  // 2) Conv stack (implicit-GEMM WMMA bf16)
  conv_wmma< 3, 66, 200, 5, 5, 2, 24, 31, 98,  96, 2, true ><<<dim3(190, 1, NIMG), 32, 0, stream>>>(images, wp[0], cb[0], a1);
  conv_wmma<24, 31,  98, 5, 5, 2, 36, 14, 47, 608, 3, false><<<dim3( 42, 1, NIMG), 32, 0, stream>>>(a1,     wp[1], cb[1], a2);
  conv_wmma<36, 14,  47, 5, 5, 2, 48,  5, 22, 928, 3, false><<<dim3(  7, 1, NIMG), 32, 0, stream>>>(a2,     wp[2], cb[2], a3);
  conv_wmma<48,  5,  22, 3, 3, 1, 64,  3, 20, 448, 4, false><<<dim3(  4, 1, NIMG), 32, 0, stream>>>(a3,     wp[3], cb[3], a4);
  conv_wmma<64,  3,  20, 3, 3, 1, 64,  1, 18, 576, 4, false><<<dim3(  2, 1, NIMG), 32, 0, stream>>>(a4,     wp[4], cb[4], a5);

  // 3) Mean over views -> feats (bf16)
  feats_mean<<<(256 * 1152 + 255) / 256, 256, 0, stream>>>(a5, feats);

  // 4) Parallel x-part precompute for CfC layer 1: P = feats * Wt  (WMMA)
  gemm_bf16_tn<<<dim3(12, 16), 32, 0, stream>>>(feats, wprnn, P, 1152, 192);

  // 5) Serial CfC scan + output head
  RnnArgs ra;
  for (int m = 0; m < 4; ++m) {
    ra.w1[m] = (const float*)d_in[11 + 2*m]; ra.b1[m] = (const float*)d_in[12 + 2*m];
    ra.w2[m] = (const float*)d_in[19 + 2*m]; ra.b2[m] = (const float*)d_in[20 + 2*m];
    ra.w3[m] = (const float*)d_in[27 + 2*m]; ra.b3[m] = (const float*)d_in[28 + 2*m];
  }
  ra.m1 = (const int*)d_in[37]; ra.m2 = (const int*)d_in[38]; ra.m3 = (const int*)d_in[39];
  ra.ow = (const float*)d_in[35]; ra.ob = (const float*)d_in[36];
  ra.P = P; ra.out = (float*)d_out;
  rnn_serial<<<1, 640, 0, stream>>>(ra);
}
